// SwinTransformerBlock_68247030334218
// MI455X (gfx1250) — matrix operand order repaired and checked by reading, not compiled, not run on hardware
//
#include <hip/hip_runtime.h>
#include <math.h>

typedef __bf16 bf16_t;
typedef bf16_t v16bf __attribute__((ext_vector_type(16)));
typedef bf16_t v8bf  __attribute__((ext_vector_type(8)));
typedef float  v8f   __attribute__((ext_vector_type(8)));

#define C_DIM   384
#define HID_DIM 1536
#define NHEAD   12
#define HDIM    32
#define NTOK    49
#define MTOT    100352
#define KC      384                      // K-chunk staged in LDS (64x384 bf16 = 48 KB)
#define SCALE_Q 0.17677669529663687f

__device__ __forceinline__ bf16_t f2bf(float f) {
  unsigned u = __builtin_bit_cast(unsigned, f);
  u += 0x7fffu + ((u >> 16) & 1u);            // round-to-nearest-even
  unsigned short s = (unsigned short)(u >> 16);
  return __builtin_bit_cast(bf16_t, s);
}

__device__ __forceinline__ v16bf cat8(v8bf lo, v8bf hi) {
  return __builtin_shufflevector(lo, hi, 0,1,2,3,4,5,6,7,8,9,10,11,12,13,14,15);
}

__device__ __forceinline__ v8f wmma_bf16(v16bf a, v16bf b, v8f c) {
  return __builtin_amdgcn_wmma_f32_16x16x32_bf16(false, a, false, b, (short)0, c,
                                                 false, false);
}

// ---------------------------------------------------------------- weights cvt
__global__ __launch_bounds__(256) void cvt_kernel(const float* __restrict__ in,
                                                  bf16_t* __restrict__ out, int n) {
  for (int i = blockIdx.x * 256 + threadIdx.x; i < n; i += gridDim.x * 256)
    out[i] = f2bf(in[i]);
}

// --------------------------------------------------- LayerNorm (+shift+window)
// mode 0: LN1, write to shifted-window row order.  mode 1: LN2, token order.
__global__ __launch_bounds__(256) void ln_kernel(const float* __restrict__ x,
                                                 const float* __restrict__ g,
                                                 const float* __restrict__ b,
                                                 bf16_t* __restrict__ out, int mode) {
  const int lane = threadIdx.x & 31, wave = threadIdx.x >> 5;
  const int token = blockIdx.x * 8 + wave;                     // < 100352
  const float* row = x + (size_t)token * C_DIM;
  float vals[12], s = 0.f;
#pragma unroll
  for (int i = 0; i < 12; ++i) { vals[i] = row[lane + 32 * i]; s += vals[i]; }
#pragma unroll
  for (int off = 16; off; off >>= 1) s += __shfl_xor(s, off, 32);
  const float mean = s * (1.f / 384.f);
  float sq = 0.f;
#pragma unroll
  for (int i = 0; i < 12; ++i) { float d = vals[i] - mean; sq += d * d; }
#pragma unroll
  for (int off = 16; off; off >>= 1) sq += __shfl_xor(sq, off, 32);
  const float rstd = rsqrtf(sq * (1.f / 384.f) + 1e-5f);

  size_t orow;
  if (mode == 0) {
    int bi = token / 3136, t = token - bi * 3136;
    int y = t / 56, xx = t - y * 56;
    int gy = (y + 53) % 56, gx = (xx + 53) % 56;               // roll(-3,-3)
    int wy = gy / 7, py = gy - wy * 7, wx = gx / 7, px = gx - wx * 7;
    orow = (size_t)(bi * 64 + wy * 8 + wx) * NTOK + py * 7 + px;
  } else {
    orow = (size_t)token;
  }
  bf16_t* o = out + orow * C_DIM;
#pragma unroll
  for (int i = 0; i < 12; ++i) {
    int c = lane + 32 * i;
    o[c] = f2bf((vals[i] - mean) * rstd * g[c] + b[c]);
  }
}

// ------------------------------------------------------------- WMMA GEMM core
// Block: 256 threads = 8 waves. Block tile 256(M) x 64(N).
// Each wave: 32x64 strip (2 M-tiles x 4 N-tiles = 8 accumulators).
// B tile (64 x KC) staged in LDS, shared by all waves; A streamed from global.
__device__ __forceinline__ void gemm_block(const bf16_t* __restrict__ A, int lda,
                                           const bf16_t* __restrict__ W, int ldw,
                                           int K, int mblk, int nbase,
                                           bf16_t* __restrict__ sB,
                                           int tid, v8f acc[2][4]) {
#pragma unroll
  for (int mt = 0; mt < 2; ++mt)
#pragma unroll
    for (int j = 0; j < 4; ++j)
#pragma unroll
      for (int e = 0; e < 8; ++e) acc[mt][j][e] = 0.f;

  const int lane = tid & 31, wave = tid >> 5;
  const int kb   = (lane >> 4) << 3;      // 0 or 8
  const int koff = (lane >> 4) << 4;      // 0 or 16
  const int m0   = mblk + wave * 32 + (lane & 15);
  const bf16_t* arow0 = A + (size_t)m0 * lda + kb;
  const bf16_t* arow1 = arow0 + (size_t)16 * lda;

  for (int kc = 0; kc < K; kc += KC) {
    __syncthreads();                       // previous chunk fully consumed
    // cooperative B-tile load: 64 rows x KC cols (16B units)
    for (int u = tid; u < 64 * (KC / 8); u += 256) {
      const int n = u / (KC / 8);
      const int kk = (u - n * (KC / 8)) * 8;
      *(v8bf*)(sB + n * KC + kk) =
          *(const v8bf*)(W + (size_t)(nbase + n) * ldw + kc + kk);
    }
    __syncthreads();
#pragma unroll
    for (int k0 = 0; k0 < KC; k0 += 32) {
      const int kg = kc + k0;
      __builtin_prefetch(arow0 + kg + 256, 0, 3);
      v16bf a0 = cat8(*(const v8bf*)(arow0 + kg),
                      *(const v8bf*)(arow0 + kg + 16));
      v16bf a1 = cat8(*(const v8bf*)(arow1 + kg),
                      *(const v8bf*)(arow1 + kg + 16));
#pragma unroll
      for (int j = 0; j < 4; ++j) {
        v16bf b = cat8(*(const v8bf*)(sB + (j * 16 + (lane & 15)) * KC + k0 + kb),
                       *(const v8bf*)(sB + (j * 16 + (lane & 15)) * KC + k0 + kb + 16));
        acc[0][j] = wmma_bf16(a0, b, acc[0][j]);
        acc[1][j] = wmma_bf16(a1, b, acc[1][j]);
      }
    }
  }
}

// ----------------------------------------------------------------- GEMM: QKV
__global__ __launch_bounds__(256) void gemm_qkv_kernel(
    const bf16_t* __restrict__ A, const bf16_t* __restrict__ W,
    const float* __restrict__ bias, bf16_t* __restrict__ qb,
    bf16_t* __restrict__ kb_, bf16_t* __restrict__ vb) {
  __shared__ __align__(64) bf16_t sB[64 * KC];
  const int tid = threadIdx.x, lane = tid & 31, wave = tid >> 5;
  const int mblk = blockIdx.x * 256;
  const int nbase = blockIdx.y * 64;
  v8f acc[2][4];
  gemm_block(A, C_DIM, W, C_DIM, C_DIM, mblk, nbase, sB, tid, acc);
  const int mloc = (lane >> 4) << 3;
#pragma unroll
  for (int j = 0; j < 4; ++j) {
    const int n = nbase + j * 16 + (lane & 15);
    const int which = n / C_DIM;
    const int hn = n - which * C_DIM;
    const int head = hn >> 5, hd = hn & 31;
    const float bn = bias[n];
#pragma unroll
    for (int mt = 0; mt < 2; ++mt)
#pragma unroll
      for (int v = 0; v < 8; ++v) {
        const int m = mblk + wave * 32 + mt * 16 + v + mloc;
        float val = acc[mt][j][v] + bn;
        const int win = m / NTOK, tok = m - win * NTOK;
        const size_t idx = (((size_t)win * NHEAD + head) * NTOK + tok) * HDIM + hd;
        if (which == 0)      qb[idx]  = f2bf(val * SCALE_Q);
        else if (which == 1) kb_[idx] = f2bf(val);
        else                 vb[idx]  = f2bf(val);
      }
  }
}

// --------------------------------------------------------------- attention
__device__ __forceinline__ int region(int g) { return g < 49 ? 0 : (g < 53 ? 1 : 2); }

__global__ __launch_bounds__(128) void attn_kernel(
    const bf16_t* __restrict__ Q, const bf16_t* __restrict__ Kb,
    const bf16_t* __restrict__ Vb, const float* __restrict__ rpb,
    bf16_t* __restrict__ AT) {
  __shared__ __align__(64) bf16_t sQ[64 * 32];
  __shared__ __align__(64) bf16_t sK[64 * 32];
  __shared__ __align__(64) bf16_t sVT[32 * 64];
  __shared__ __align__(64) bf16_t sP[64 * 64];

  const int win = blockIdx.x, head = blockIdx.y;
  const int tid = threadIdx.x, lane = tid & 31, wave = tid >> 5;
  const size_t base = ((size_t)win * NHEAD + head) * NTOK * HDIM;

  const bf16_t z = f2bf(0.f);
  for (int i = tid; i < 64 * 32; i += 128) {
    const int r = i >> 5, c = i & 31;
    const bool ok = r < NTOK;
    const size_t gi = base + (size_t)r * HDIM + c;
    sQ[i] = ok ? Q[gi] : z;
    sK[i] = ok ? Kb[gi] : z;
    sVT[c * 64 + r] = ok ? Vb[gi] : z;
  }
  __syncthreads();

  // S = Q * K^T : one 16-row tile per wave, 4 column tiles, K = 32
  v8f s[4];
#pragma unroll
  for (int j = 0; j < 4; ++j)
#pragma unroll
    for (int e = 0; e < 8; ++e) s[j][e] = 0.f;
  {
    const int m = wave * 16 + (lane & 15);
    const int kb = (lane >> 4) << 3;
    v16bf a = cat8(*(const v8bf*)(sQ + m * 32 + kb),
                   *(const v8bf*)(sQ + m * 32 + kb + 16));
#pragma unroll
    for (int j = 0; j < 4; ++j) {
      const int n = j * 16 + (lane & 15);
      v16bf b = cat8(*(const v8bf*)(sK + n * 32 + kb),
                     *(const v8bf*)(sK + n * 32 + kb + 16));
      s[j] = wmma_bf16(a, b, s[j]);
    }
  }

  // relative-position bias + shifted-window mask
  const int wi = win & 63, wy = wi >> 3, wx = wi & 7;
  const int mloc = (lane >> 4) << 3;
#pragma unroll
  for (int j = 0; j < 4; ++j) {
    const int jc = j * 16 + (lane & 15);
    const int yj = jc / 7, xj = jc - yj * 7;
    const int labj = (jc < NTOK) ? region(wy * 7 + yj) * 3 + region(wx * 7 + xj) : -1;
#pragma unroll
    for (int v = 0; v < 8; ++v) {
      const int i = wave * 16 + v + mloc;
      float sv;
      if (jc >= NTOK)      sv = -1e30f;          // padded key columns
      else if (i >= NTOK)  sv = 0.f;             // padded rows (discarded)
      else {
        const int yi = i / 7, xi = i - yi * 7;
        const int labi = region(wy * 7 + yi) * 3 + region(wx * 7 + xi);
        const float bias = rpb[((yi - yj + 6) * 13 + (xi - xj + 6)) * NHEAD + head];
        sv = s[j][v] + bias + (labi == labj ? 0.f : -100.f);
      }
      s[j][v] = sv;
    }
  }

  // row softmax (rows split by lane half -> width-16 xor reductions)
#pragma unroll
  for (int v = 0; v < 8; ++v) {
    float mx = fmaxf(fmaxf(s[0][v], s[1][v]), fmaxf(s[2][v], s[3][v]));
#pragma unroll
    for (int off = 8; off; off >>= 1) mx = fmaxf(mx, __shfl_xor(mx, off, 16));
    float sum = 0.f;
#pragma unroll
    for (int j = 0; j < 4; ++j) { float e = __expf(s[j][v] - mx); s[j][v] = e; sum += e; }
#pragma unroll
    for (int off = 8; off; off >>= 1) sum += __shfl_xor(sum, off, 16);
    const float inv = 1.f / sum;
    const int i = wave * 16 + v + mloc;
#pragma unroll
    for (int j = 0; j < 4; ++j)
      sP[i * 64 + j * 16 + (lane & 15)] = f2bf(s[j][v] * inv);
  }
  __syncthreads();

  // O = P * V : 16x32 per wave, K = 64 (2 steps)
  v8f o[2];
#pragma unroll
  for (int jt = 0; jt < 2; ++jt)
#pragma unroll
    for (int e = 0; e < 8; ++e) o[jt][e] = 0.f;
#pragma unroll
  for (int ks = 0; ks < 2; ++ks) {
    const int k0 = ks * 32;
    const int m = wave * 16 + (lane & 15);
    const int kb = (lane >> 4) << 3;
    v16bf a = cat8(*(const v8bf*)(sP + m * 64 + k0 + kb),
                   *(const v8bf*)(sP + m * 64 + k0 + kb + 16));
#pragma unroll
    for (int jt = 0; jt < 2; ++jt) {
      const int n = jt * 16 + (lane & 15);
      v16bf b = cat8(*(const v8bf*)(sVT + n * 64 + k0 + kb),
                     *(const v8bf*)(sVT + n * 64 + k0 + kb + 16));
      o[jt] = wmma_bf16(a, b, o[jt]);
    }
  }
#pragma unroll
  for (int jt = 0; jt < 2; ++jt)
#pragma unroll
    for (int v = 0; v < 8; ++v) {
      const int i = wave * 16 + v + mloc;
      if (i < NTOK) {
        const int col = jt * 16 + (lane & 15);
        AT[((size_t)win * NTOK + i) * C_DIM + head * HDIM + col] = f2bf(o[jt][v]);
      }
    }
}

// ------------------------------------------ GEMM: proj (+reverse+roll+resid)
__global__ __launch_bounds__(256) void gemm_proj_kernel(
    const bf16_t* __restrict__ A, const bf16_t* __restrict__ W,
    const float* __restrict__ bias, const float* __restrict__ x_in,
    float* __restrict__ xout) {
  __shared__ __align__(64) bf16_t sB[64 * KC];
  const int tid = threadIdx.x, lane = tid & 31, wave = tid >> 5;
  const int mblk = blockIdx.x * 256;
  const int nbase = blockIdx.y * 64;
  v8f acc[2][4];
  gemm_block(A, C_DIM, W, C_DIM, C_DIM, mblk, nbase, sB, tid, acc);
  const int mloc = (lane >> 4) << 3;
#pragma unroll
  for (int j = 0; j < 4; ++j) {
    const int n = nbase + j * 16 + (lane & 15);
    const float bn = bias[n];
#pragma unroll
    for (int mt = 0; mt < 2; ++mt)
#pragma unroll
      for (int v = 0; v < 8; ++v) {
        const int m = mblk + wave * 32 + mt * 16 + v + mloc;
        const int win = m / NTOK, tok = m - win * NTOK;
        const int bi = win >> 6, wg = win & 63, wy = wg >> 3, wx = wg & 7;
        const int py = tok / 7, px = tok - py * 7;
        int y0 = wy * 7 + py + 3; if (y0 >= 56) y0 -= 56;   // roll(+3,+3)
        int x0 = wx * 7 + px + 3; if (x0 >= 56) x0 -= 56;
        const size_t t = ((size_t)bi * 3136 + y0 * 56 + x0) * C_DIM + n;
        xout[t] = x_in[t] + acc[mt][j][v] + bn;
      }
  }
}

// ---------------------------------------------------------- GEMM: fc1 + GELU
__global__ __launch_bounds__(256) void gemm_fc1_kernel(
    const bf16_t* __restrict__ A, const bf16_t* __restrict__ W,
    const float* __restrict__ bias, bf16_t* __restrict__ h1) {
  __shared__ __align__(64) bf16_t sB[64 * KC];
  const int tid = threadIdx.x, lane = tid & 31, wave = tid >> 5;
  const int mblk = blockIdx.x * 256;
  const int nbase = blockIdx.y * 64;
  v8f acc[2][4];
  gemm_block(A, C_DIM, W, C_DIM, C_DIM, mblk, nbase, sB, tid, acc);
  const int mloc = (lane >> 4) << 3;
#pragma unroll
  for (int j = 0; j < 4; ++j) {
    const int n = nbase + j * 16 + (lane & 15);
    const float bn = bias[n];
#pragma unroll
    for (int mt = 0; mt < 2; ++mt)
#pragma unroll
      for (int v = 0; v < 8; ++v) {
        const int m = mblk + wave * 32 + mt * 16 + v + mloc;
        const float t = acc[mt][j][v] + bn;
        const float gl = 0.5f * t * (1.f + erff(t * 0.70710678118654752f));
        h1[(size_t)m * HID_DIM + n] = f2bf(gl);
      }
  }
}

// ------------------------------------------------------ GEMM: fc2 + residual
__global__ __launch_bounds__(256) void gemm_fc2_kernel(
    const bf16_t* __restrict__ A, const bf16_t* __restrict__ W,
    const float* __restrict__ bias, float* __restrict__ xout) {
  __shared__ __align__(64) bf16_t sB[64 * KC];
  const int tid = threadIdx.x, lane = tid & 31, wave = tid >> 5;
  const int mblk = blockIdx.x * 256;
  const int nbase = blockIdx.y * 64;
  v8f acc[2][4];
  gemm_block(A, HID_DIM, W, HID_DIM, HID_DIM, mblk, nbase, sB, tid, acc);
  const int mloc = (lane >> 4) << 3;
#pragma unroll
  for (int j = 0; j < 4; ++j) {
    const int n = nbase + j * 16 + (lane & 15);
    const float bn = bias[n];
#pragma unroll
    for (int mt = 0; mt < 2; ++mt)
#pragma unroll
      for (int v = 0; v < 8; ++v) {
        const int m = mblk + wave * 32 + mt * 16 + v + mloc;
        float* p = xout + (size_t)m * C_DIM + n;
        *p = *p + acc[mt][j][v] + bn;
      }
  }
}

// ------------------------------------------------------------------ launcher
extern "C" void kernel_launch(void* const* d_in, const int* in_sizes, int n_in,
                              void* d_out, int out_size, void* d_ws, size_t ws_size,
                              hipStream_t stream) {
  (void)in_sizes; (void)n_in; (void)out_size; (void)ws_size;
  const float* x       = (const float*)d_in[0];
  const float* n1g     = (const float*)d_in[1];
  const float* n1b     = (const float*)d_in[2];
  const float* qkv_w   = (const float*)d_in[3];
  const float* qkv_b   = (const float*)d_in[4];
  const float* rpb     = (const float*)d_in[5];
  const float* proj_w  = (const float*)d_in[6];
  const float* proj_b  = (const float*)d_in[7];
  const float* n2g     = (const float*)d_in[8];
  const float* n2b     = (const float*)d_in[9];
  const float* fc1_w   = (const float*)d_in[10];
  const float* fc1_b   = (const float*)d_in[11];
  const float* fc2_w   = (const float*)d_in[12];
  const float* fc2_b   = (const float*)d_in[13];
  float* xout = (float*)d_out;   // holds x_new, then final output (accumulated)

  char* ws = (char*)d_ws;
  const size_t SZ_ACT   = (size_t)MTOT * C_DIM * 2;          // 77,070,336 B
  const size_t OFF_WQKV = 0;
  const size_t OFF_WPRJ = OFF_WQKV + (size_t)1152 * 384 * 2;
  const size_t OFF_WFC1 = OFF_WPRJ + (size_t)384 * 384 * 2;
  const size_t OFF_WFC2 = OFF_WFC1 + (size_t)1536 * 384 * 2;
  const size_t OFF_XW   = 4u << 20;                          // 4 MB
  const size_t OFF_Q    = OFF_XW + SZ_ACT;
  const size_t OFF_K    = OFF_Q + SZ_ACT;
  const size_t OFF_V    = OFF_K + SZ_ACT;
  const size_t OFF_AT   = OFF_V + SZ_ACT;
  bf16_t* wQKV = (bf16_t*)(ws + OFF_WQKV);
  bf16_t* wPRJ = (bf16_t*)(ws + OFF_WPRJ);
  bf16_t* wFC1 = (bf16_t*)(ws + OFF_WFC1);
  bf16_t* wFC2 = (bf16_t*)(ws + OFF_WFC2);
  bf16_t* XW   = (bf16_t*)(ws + OFF_XW);    // also reused as y_ln
  bf16_t* Qb   = (bf16_t*)(ws + OFF_Q);     // q/k/v/attn region reused as h1
  bf16_t* Kb   = (bf16_t*)(ws + OFF_K);
  bf16_t* Vb   = (bf16_t*)(ws + OFF_V);
  bf16_t* AT   = (bf16_t*)(ws + OFF_AT);
  bf16_t* H1   = Qb;
  bf16_t* YLN  = XW;

  // 1) weights -> bf16
  cvt_kernel<<<432, 256, 0, stream>>>(qkv_w, wQKV, 1152 * 384);
  cvt_kernel<<<144, 256, 0, stream>>>(proj_w, wPRJ, 384 * 384);
  cvt_kernel<<<576, 256, 0, stream>>>(fc1_w, wFC1, 1536 * 384);
  cvt_kernel<<<576, 256, 0, stream>>>(fc2_w, wFC2, 384 * 1536);

  // 2) LN1 + roll(-3) + window partition
  ln_kernel<<<MTOT / 8, 256, 0, stream>>>(x, n1g, n1b, XW, 0);

  // 3) QKV GEMM (M=100352, N=1152, K=384)
  gemm_qkv_kernel<<<dim3(MTOT / 256, 1152 / 64), 256, 0, stream>>>(
      XW, wQKV, qkv_b, Qb, Kb, Vb);

  // 4) windowed attention: one block per (window, head)
  attn_kernel<<<dim3(2048, NHEAD), 128, 0, stream>>>(Qb, Kb, Vb, rpb, AT);

  // 5) proj GEMM + window reverse + roll(+3) + residual -> x_new in d_out
  gemm_proj_kernel<<<dim3(MTOT / 256, 384 / 64), 256, 0, stream>>>(
      AT, wPRJ, proj_b, x, xout);

  // 6) LN2 (token order)
  ln_kernel<<<MTOT / 8, 256, 0, stream>>>(xout, n2g, n2b, YLN, 1);

  // 7) fc1 + GELU (M=100352, N=1536, K=384)
  gemm_fc1_kernel<<<dim3(MTOT / 256, 1536 / 64), 256, 0, stream>>>(
      YLN, wFC1, fc1_b, H1);

  // 8) fc2 + residual accumulate into d_out (M=100352, N=384, K=1536)
  gemm_fc2_kernel<<<dim3(MTOT / 256, 384 / 64), 256, 0, stream>>>(
      H1, wFC2, fc2_b, xout);
}
